// MultiHeadSelfAttention_29351806501634
// MI455X (gfx1250) — matrix operand order repaired and checked by reading, not compiled, not run on hardware
//
#include <hip/hip_runtime.h>

typedef __attribute__((ext_vector_type(16))) _Float16 v16h;
typedef __attribute__((ext_vector_type(8)))  _Float16 v8h;
typedef __attribute__((ext_vector_type(4)))  _Float16 v4h;
typedef __attribute__((ext_vector_type(8)))  float    v8f;
typedef __attribute__((address_space(3)))    _Float16 lds_f16_t;

#define B_   8
#define N_   2048
#define DIN_ 256
#define DM_  256
#define NH_  4
#define DK_  64

__device__ __forceinline__ v16h combine16(v8h lo, v8h hi) {
    v16h r;
#pragma unroll
    for (int i = 0; i < 8; ++i) { r[i] = lo[i]; r[i + 8] = hi[i]; }
    return r;
}
// A-fragment (16x32 MxK): lane holds row M=lane&15; K = hk*16 + i (16 contiguous)
__device__ __forceinline__ v16h load_a_frag(const _Float16* rowp, int hk) {
    return combine16(*(const v8h*)(rowp + hk * 16),
                     *(const v8h*)(rowp + hk * 16 + 8));
}
// B-fragment (32x16 KxN): lane holds col N=lane&15; K = hk*16 + i (16 contiguous)
__device__ __forceinline__ v16h load_b_frag(const _Float16* p) {
    return combine16(*(const v8h*)(p), *(const v8h*)(p + 8));
}
__device__ __forceinline__ v8f wmma_f16(v16h a, v16h b, v8f c) {
    return __builtin_amdgcn_wmma_f32_16x16x32_f16(false, a, false, b,
                                                  (short)0, c, false, false);
}
// async global -> LDS, 16B per lane (ASYNCcnt-tracked)
__device__ __forceinline__ void async_b128(unsigned lds_off, const _Float16* gptr) {
    asm volatile("global_load_async_to_lds_b128 %0, %1, off"
                 :: "v"(lds_off), "v"(gptr) : "memory");
}
__device__ __forceinline__ void wait_async0() {
    asm volatile("s_wait_asynccnt 0" ::: "memory");
}

// ---------------- conversions ----------------
__global__ __launch_bounds__(256) void cvt_x_kernel(
    const float* __restrict__ x, _Float16* __restrict__ x16) {
    const int idx = (blockIdx.x * 256 + threadIdx.x) * 4;
    const float4 v = *(const float4*)(x + idx);
    v4h o = { (_Float16)v.x, (_Float16)v.y, (_Float16)v.z, (_Float16)v.w };
    *(v4h*)(x16 + idx) = o;
}

// Wt[n][k] = (f16) W[k][n]  (256x256)
__global__ __launch_bounds__(256) void cvt_w_kernel(
    const float* __restrict__ W, _Float16* __restrict__ Wt) {
    const int idx = blockIdx.x * 256 + threadIdx.x;
    const int n = idx >> 8, k = idx & 255;
    Wt[idx] = (_Float16)W[k * DM_ + n];
}

// ---------------- mask start/end scan ----------------
__global__ __launch_bounds__(256) void mask_scan_kernel(
    const int* __restrict__ mask, int* __restrict__ se) {
    __shared__ int smin[256], smax[256];
    const int b = blockIdx.x, t = threadIdx.x;
    int mn = N_, mx = -1;
    for (int p = t; p < N_; p += 256) {
        if (mask[b * N_ + p]) { mn = min(mn, p); mx = max(mx, p); }
    }
    smin[t] = mn; smax[t] = mx;
    __syncthreads();
    for (int s = 128; s > 0; s >>= 1) {
        if (t < s) { smin[t] = min(smin[t], smin[t + s]); smax[t] = max(smax[t], smax[t + s]); }
        __syncthreads();
    }
    if (t == 0) { se[b * 2] = smin[0]; se[b * 2 + 1] = smax[0]; }  // end (exclusive) = last True idx
}

// ---------------- generic 16384x256 @ 256x256 GEMM ----------------
// mode 0: -> q_ws f16 [b,h,n,dk] * 1/sqrt(dk)
// mode 1: -> k_ws f16 [b,h,n,dk]
// mode 2: -> vt_ws f16 [b,h,dk,n]  (transposed V)
// mode 3: -> out  f32 [row, col]
__global__ __launch_bounds__(256) void gemm_kernel(
    const _Float16* __restrict__ A, const _Float16* __restrict__ Wt,
    const float* __restrict__ bias, void* __restrict__ out, int mode) {
    const int lane = threadIdx.x & 31;
    const int wave = threadIdx.x >> 5;
    const int hk = lane >> 4, ln16 = lane & 15;
    const int tile = blockIdx.x * 8 + wave;     // 4096 wave-tiles
    const int tm = tile >> 2;                   // row tile 0..1023
    const int cg = tile & 3;                    // col group (4 x 16 cols)
    const int row = tm * 16 + ln16;

    v8f c[4];
#pragma unroll
    for (int tc = 0; tc < 4; ++tc) {
        const float bval = bias[cg * 64 + tc * 16 + ln16];
#pragma unroll
        for (int e = 0; e < 8; ++e) c[tc][e] = bval;
    }

    for (int kc = 0; kc < DIN_ / 32; ++kc) {
        const v16h a = load_a_frag(A + row * DIN_ + kc * 32, hk);
#pragma unroll
        for (int tc = 0; tc < 4; ++tc) {
            const v16h b = load_b_frag(Wt + (cg * 64 + tc * 16 + ln16) * DIN_ +
                                       kc * 32 + hk * 16);
            c[tc] = wmma_f16(a, b, c[tc]);
        }
    }

#pragma unroll
    for (int tc = 0; tc < 4; ++tc) {
        const int col = cg * 64 + tc * 16 + ln16;
        const int h = col >> 6, d = col & 63;
#pragma unroll
        for (int e = 0; e < 8; ++e) {
            const int m = (lane < 16) ? e : e + 8;
            const int grow = tm * 16 + m;
            if (mode == 3) {
                ((float*)out)[grow * DM_ + col] = c[tc][e];
            } else {
                const int bb = grow >> 11, npos = grow & (N_ - 1);
                const int bh = bb * NH_ + h;
                _Float16* oh = (_Float16*)out;
                if (mode == 2)
                    oh[(bh * DK_ + d) * N_ + npos] = (_Float16)c[tc][e];
                else if (mode == 0)
                    oh[(bh * N_ + npos) * DK_ + d] = (_Float16)(c[tc][e] * 0.125f);
                else
                    oh[(bh * N_ + npos) * DK_ + d] = (_Float16)c[tc][e];
            }
        }
    }
}

// ---------------- flash attention (LDS-staged K/V via async DMA) ----------------
__global__ __launch_bounds__(256) void flash_attn_kernel(
    const _Float16* __restrict__ q_ws, const _Float16* __restrict__ k_ws,
    const _Float16* __restrict__ vt_ws, const int* __restrict__ se,
    _Float16* __restrict__ att_ws) {
    __shared__ _Float16 Klds[2][32 * 64];  // [key][dim], 4KB per buffer
    __shared__ _Float16 Vlds[2][64 * 32];  // [dim][key], 4KB per buffer
    __shared__ _Float16 ldsP[8][16 * 32];
    const int tid = threadIdx.x;
    const int lane = tid & 31;
    const int wave = tid >> 5;
    const int hk = lane >> 4, ln16 = lane & 15;
    const int bh = blockIdx.y;
    const int bb = bh >> 2, h = bh & 3;
    const int qbase = blockIdx.x * 128 + wave * 16;
    const int start = se[bb * 2], endx = se[bb * 2 + 1];

    const _Float16* Q  = q_ws  + (size_t)bh * N_ * DK_;
    const _Float16* K  = k_ws  + (size_t)bh * N_ * DK_;
    const _Float16* Vt = vt_ws + (size_t)bh * DK_ * N_;

    // per-thread async-staging addresses (K: 32 rows x 8 chunks; V: 64 rows x 4 chunks)
    const int krow = tid >> 3, kch = tid & 7;          // 32 x 8 chunks of 8 halfs
    const int vrow = tid >> 2, vch = tid & 3;          // 64 x 4 chunks of 8 halfs
    const unsigned lk0 = (unsigned)(uintptr_t)(lds_f16_t*)&Klds[0][krow * 64 + kch * 8];
    const unsigned lk1 = (unsigned)(uintptr_t)(lds_f16_t*)&Klds[1][krow * 64 + kch * 8];
    const unsigned lv0 = (unsigned)(uintptr_t)(lds_f16_t*)&Vlds[0][vrow * 32 + vch * 8];
    const unsigned lv1 = (unsigned)(uintptr_t)(lds_f16_t*)&Vlds[1][vrow * 32 + vch * 8];

    const int qrow = qbase + ln16;
    const v16h qf0 = load_a_frag(Q + qrow * DK_, hk);
    const v16h qf1 = load_a_frag(Q + qrow * DK_ + 32, hk);

    v8f o[4];
#pragma unroll
    for (int t = 0; t < 4; ++t)
#pragma unroll
        for (int e = 0; e < 8; ++e) o[t][e] = 0.f;

    float mrow[8], lrow[8];
    int rowpos[8]; bool qv[8];
#pragma unroll
    for (int e = 0; e < 8; ++e) {
        mrow[e] = -__builtin_inff(); lrow[e] = 0.f;
        rowpos[e] = qbase + ((lane < 16) ? e : e + 8);
        qv[e] = (rowpos[e] >= start) && (rowpos[e] < endx);
    }

    // stage first tile
    async_b128(lk0, K + krow * DK_ + kch * 8);
    async_b128(lv0, Vt + vrow * N_ + vch * 8);

    for (int kb = 0; kb < N_ / 32; ++kb) {
        wait_async0();
        __syncthreads();   // tile (kb&1) fully landed for every wave;
                           // also: all waves done reading the other buffer

        if (kb + 1 < N_ / 32) {   // stage next tile while computing this one
            const _Float16* gk = K + (kb * 32 + 32 + krow) * DK_ + kch * 8;
            const _Float16* gv = Vt + vrow * N_ + kb * 32 + 32 + vch * 8;
            async_b128((kb & 1) ? lk0 : lk1, gk);
            async_b128((kb & 1) ? lv0 : lv1, gv);
        }

        const _Float16* Kt = &Klds[kb & 1][0];
        const _Float16* Vl = &Vlds[kb & 1][0];

        float s0[8], s1[8];
#pragma unroll
        for (int sub = 0; sub < 2; ++sub) {
            const int keyl = sub * 16 + ln16;            // key within tile
            const _Float16* krp = Kt + keyl * DK_;
            const v16h bf0 = load_b_frag(krp + hk * 16);
            const v16h bf1 = load_b_frag(krp + 32 + hk * 16);
            v8f s;
#pragma unroll
            for (int e = 0; e < 8; ++e) s[e] = 0.f;
            s = wmma_f16(qf0, bf0, s);
            s = wmma_f16(qf1, bf1, s);
            const int key = kb * 32 + keyl;
            const bool kvld = (key >= start) && (key < endx);
            float* sp = sub ? s1 : s0;
#pragma unroll
            for (int e = 0; e < 8; ++e)
                sp[e] = s[e] + ((qv[e] && kvld) ? 0.f : -1e10f);
        }

        // online softmax; row reductions across 16-lane halves (xor 1,2,4,8)
#pragma unroll
        for (int e = 0; e < 8; ++e) {
            float r = fmaxf(s0[e], s1[e]);
#pragma unroll
            for (int off = 1; off < 16; off <<= 1)
                r = fmaxf(r, __shfl_xor(r, off, 32));
            const float mn = fmaxf(mrow[e], r);
            const float alpha = __expf(mrow[e] - mn);
            const float p0 = __expf(s0[e] - mn);
            const float p1 = __expf(s1[e] - mn);
            float rs = p0 + p1;
#pragma unroll
            for (int off = 1; off < 16; off <<= 1)
                rs += __shfl_xor(rs, off, 32);
            lrow[e] = lrow[e] * alpha + rs;
            mrow[e] = mn;
#pragma unroll
            for (int t = 0; t < 4; ++t) o[t][e] *= alpha;
            s0[e] = p0; s1[e] = p1;
        }

        // D-layout P -> LDS -> A-fragment
#pragma unroll
        for (int e = 0; e < 8; ++e) {
            const int r = (lane < 16) ? e : e + 8;
            ldsP[wave][r * 32 + ln16]      = (_Float16)s0[e];
            ldsP[wave][r * 32 + 16 + ln16] = (_Float16)s1[e];
        }
        asm volatile("s_wait_dscnt 0" ::: "memory");
        const v16h pa = combine16(*(const v8h*)(&ldsP[wave][ln16 * 32 + hk * 16]),
                                  *(const v8h*)(&ldsP[wave][ln16 * 32 + hk * 16 + 8]));

#pragma unroll
        for (int t = 0; t < 4; ++t) {
            const v16h vb = load_b_frag(Vl + (t * 16 + ln16) * 32 + hk * 16);
            o[t] = wmma_f16(pa, vb, o[t]);
        }
    }

    // normalize + write att in [b, n, h*dk] f16
#pragma unroll
    for (int e = 0; e < 8; ++e) {
        const float inv = (lrow[e] > 0.f) ? (1.0f / lrow[e]) : 0.f;
        const int base = (bb * N_ + rowpos[e]) * DM_ + h * DK_;
#pragma unroll
        for (int t = 0; t < 4; ++t)
            att_ws[base + t * 16 + ln16] = (_Float16)(o[t][e] * inv);
    }
}

extern "C" void kernel_launch(void* const* d_in, const int* in_sizes, int n_in,
                              void* d_out, int out_size, void* d_ws, size_t ws_size,
                              hipStream_t stream) {
    const float* x    = (const float*)d_in[0];
    const int*   mask = (const int*)d_in[1];
    const float* Wq = (const float*)d_in[2];
    const float* bq = (const float*)d_in[3];
    const float* Wk = (const float*)d_in[4];
    const float* bk = (const float*)d_in[5];
    const float* Wv = (const float*)d_in[6];
    const float* bv = (const float*)d_in[7];
    const float* Wo = (const float*)d_in[8];
    const float* bo = (const float*)d_in[9];
    float* out = (float*)d_out;

    char* ws = (char*)d_ws;
    const size_t MB = 1024 * 1024;
    _Float16* x16    = (_Float16*)(ws);             // 8 MB (16384x256)
    _Float16* q_ws   = (_Float16*)(ws + 8 * MB);    // 8 MB
    _Float16* k_ws   = (_Float16*)(ws + 16 * MB);   // 8 MB
    _Float16* vt_ws  = (_Float16*)(ws + 24 * MB);   // 8 MB
    _Float16* att_ws = (_Float16*)(ws + 32 * MB);   // 8 MB
    _Float16* wt_q   = (_Float16*)(ws + 40 * MB);   // 128 KB each
    _Float16* wt_k   = wt_q + 65536;
    _Float16* wt_v   = wt_k + 65536;
    _Float16* wt_o   = wt_v + 65536;
    int*      se_ws  = (int*)(wt_o + 65536);

    cvt_x_kernel<<<4096, 256, 0, stream>>>(x, x16);
    cvt_w_kernel<<<256, 256, 0, stream>>>(Wq, wt_q);
    cvt_w_kernel<<<256, 256, 0, stream>>>(Wk, wt_k);
    cvt_w_kernel<<<256, 256, 0, stream>>>(Wv, wt_v);
    cvt_w_kernel<<<256, 256, 0, stream>>>(Wo, wt_o);
    mask_scan_kernel<<<B_, 256, 0, stream>>>(mask, se_ws);

    gemm_kernel<<<512, 256, 0, stream>>>(x16, wt_q, bq, q_ws, 0);
    gemm_kernel<<<512, 256, 0, stream>>>(x16, wt_k, bk, k_ws, 1);
    gemm_kernel<<<512, 256, 0, stream>>>(x16, wt_v, bv, vt_ws, 2);

    flash_attn_kernel<<<dim3(N_ / 128, B_ * NH_), 256, 0, stream>>>(
        q_ws, k_ws, vt_ws, se_ws, att_ws);

    gemm_kernel<<<512, 256, 0, stream>>>(att_ws, wt_o, bo, out, 3);
}
